// TorchairDeepseekV2Model_11596411699305
// MI455X (gfx1250) — matrix operand order repaired and checked by reading, not compiled, not run on hardware
//
#include <hip/hip_runtime.h>

// ---------------------------------------------------------------------------
// Problem constants (match the reference)
// ---------------------------------------------------------------------------
static constexpr int B_   = 2;
static constexpr int S_   = 2048;
static constexpr int T_   = B_ * S_;      // 4096
static constexpr int HID  = 2048;
static constexpr int H_   = 16;
static constexpr int DN   = 128;
static constexpr int DR   = 64;
static constexpr int DV   = 128;
static constexpr int QKD  = DN + DR;      // 192
static constexpr int QLR  = 1536;
static constexpr int KVR  = 512;
static constexpr float EPS   = 1e-6f;
static constexpr float SCALE_ = 0.07216878364870323f; // 192^-0.5

// ---------------------------------------------------------------------------
// Vector types / WMMA helpers
// ---------------------------------------------------------------------------
typedef __attribute__((ext_vector_type(16))) __bf16 v16bf;
typedef __attribute__((ext_vector_type(8)))  __bf16 v8bf;
typedef __attribute__((ext_vector_type(8)))  float  v8f;
typedef __attribute__((ext_vector_type(4)))  int    v4i;

typedef __attribute__((address_space(1))) v4i as1_v4i;
typedef __attribute__((address_space(3))) v4i as3_v4i;

__device__ __forceinline__ v8f wmma_bf16(v16bf a, v16bf b, v8f c) {
  return __builtin_amdgcn_wmma_f32_16x16x32_bf16(
      false, a, false, b, (short)0, c, false, false);
}

__device__ __forceinline__ __bf16 to_bf16(float f) {
  unsigned u = __builtin_bit_cast(unsigned, f);
  unsigned r = u + 0x7FFFu + ((u >> 16) & 1u);   // round-to-nearest-even
  unsigned short h = (unsigned short)(r >> 16);
  return __builtin_bit_cast(__bf16, h);
}

union frag { v16bf v; v8bf h2[2]; };

// 16-byte global -> LDS copy; async (ASYNCcnt) when the toolchain exposes it
__device__ __forceinline__ void lds_cp16(__bf16* dst, const __bf16* src) {
#if __has_builtin(__builtin_amdgcn_global_load_async_to_lds_b128)
  __builtin_amdgcn_global_load_async_to_lds_b128(
      (as1_v4i*)src, (as3_v4i*)dst, 0, 0);
#else
  *(v8bf*)dst = *(const v8bf*)src;
#endif
}
__device__ __forceinline__ void lds_cp_wait() {
#if __has_builtin(__builtin_amdgcn_global_load_async_to_lds_b128)
#if __has_builtin(__builtin_amdgcn_s_wait_asynccnt)
  __builtin_amdgcn_s_wait_asynccnt(0);
#else
  asm volatile("s_wait_asynccnt 0" ::: "memory");
#endif
#endif
}

// ---------------------------------------------------------------------------
// fp32 -> bf16 conversion (grid-stride)
// ---------------------------------------------------------------------------
__global__ void cvt_bf16_kernel(const float* __restrict__ src,
                                __bf16* __restrict__ dst, long n) {
  long i = (long)blockIdx.x * blockDim.x + threadIdx.x;
  long stride = (long)gridDim.x * blockDim.x;
  for (; i < n; i += stride) dst[i] = to_bf16(src[i]);
}

// ---------------------------------------------------------------------------
// fp32 [K][N] -> bf16 transposed [N][K] (32x32 LDS tile transpose)
// grid = (N/32, K/32), block = 256
// ---------------------------------------------------------------------------
__global__ __launch_bounds__(256)
void cvtT_bf16_kernel(const float* __restrict__ src, __bf16* __restrict__ dst,
                      int K, int N) {
  __shared__ __bf16 tile[32][33];
  const int kt = blockIdx.y * 32, nt = blockIdx.x * 32;
  const int tx = threadIdx.x & 31, ty = threadIdx.x >> 5;  // 32 x 8
  #pragma unroll
  for (int i = 0; i < 4; ++i) {
    const int k = ty + i * 8;
    tile[tx][k] = to_bf16(src[(long)(kt + k) * N + nt + tx]);
  }
  __syncthreads();
  #pragma unroll
  for (int i = 0; i < 4; ++i) {
    const int n = ty + i * 8;
    dst[(long)(nt + n) * K + kt + tx] = tile[n][tx];
  }
}

// ---------------------------------------------------------------------------
// bf16 GEMM: C[M,N] f32 = A[M,K] bf16 (row-major) * Bt[N,K] bf16 (pre-transposed).
// Workgroup tile 128x64, 8 waves, 32x32 per wave (4 WMMAs/K-step).
// Double-buffered async global->LDS staging, one barrier per K-step.
// Fragment layouts (16-bit, wave32):
//   A: lane m = l&15, VGPR K-runs at k = hi*8 and 16+hi*8   (2x b128)
//   B: lane n = l&15, VGPR K-runs at k = hi*8 and 16+hi*8   (2x b128)
// ---------------------------------------------------------------------------
static constexpr int LDT = 40; // LDS row stride in bf16 elems (80B, 16B-aligned)

__global__ __launch_bounds__(256)
void gemm_bf16_kernel(const __bf16* __restrict__ A, const __bf16* __restrict__ Bt,
                      float* __restrict__ C, int M, int N, int K) {
  __shared__ __attribute__((aligned(16))) __bf16 lA[2][128 * LDT];
  __shared__ __attribute__((aligned(16))) __bf16 lB[2][64 * LDT]; // [n][k]

  const int tid  = threadIdx.x;
  const int wave = tid >> 5;
  const int lane = tid & 31;
  const int lm   = lane & 15;
  const int hi   = lane >> 4;

  const int m0 = blockIdx.y * 128;
  const int n0 = blockIdx.x * 64;
  const int wm = (wave >> 1) * 32;
  const int wn = (wave & 1) * 32;

  v8f acc00 = {}, acc01 = {}, acc10 = {}, acc11 = {};

  const int ar = tid >> 1, ac = (tid & 1) * 16;  // A loader: 128 rows x 32
  const int bn = tid >> 2, bk = (tid & 3) * 8;   // B loader: 64 rows x 32

  const __bf16* Arow = A  + (long)(m0 + ar) * K;
  const __bf16* Brow = Bt + (long)(n0 + bn) * K;

  auto stage = [&](int buf, int kt) {
    lds_cp16(&lA[buf][ar * LDT + ac],     Arow + kt + ac);
    lds_cp16(&lA[buf][ar * LDT + ac + 8], Arow + kt + ac + 8);
    lds_cp16(&lB[buf][bn * LDT + bk],     Brow + kt + bk);
  };

  const int nt = K >> 5;
  stage(0, 0);

  for (int t = 0; t < nt; ++t) {
    lds_cp_wait();
    __syncthreads();                // tile t landed; tile t-1 reads retired
    if (t + 1 < nt) stage((t + 1) & 1, (t + 1) << 5);

    const __bf16* bA = &lA[t & 1][0];
    const __bf16* bB = &lB[t & 1][0];
    frag a0, a1, b0, b1;
    a0.h2[0] = *(const v8bf*)&bA[(wm + lm) * LDT + hi * 8];
    a0.h2[1] = *(const v8bf*)&bA[(wm + lm) * LDT + 16 + hi * 8];
    a1.h2[0] = *(const v8bf*)&bA[(wm + 16 + lm) * LDT + hi * 8];
    a1.h2[1] = *(const v8bf*)&bA[(wm + 16 + lm) * LDT + 16 + hi * 8];
    b0.h2[0] = *(const v8bf*)&bB[(wn + lm) * LDT + hi * 8];
    b0.h2[1] = *(const v8bf*)&bB[(wn + lm) * LDT + 16 + hi * 8];
    b1.h2[0] = *(const v8bf*)&bB[(wn + 16 + lm) * LDT + hi * 8];
    b1.h2[1] = *(const v8bf*)&bB[(wn + 16 + lm) * LDT + 16 + hi * 8];

    acc00 = wmma_bf16(a0.v, b0.v, acc00);
    acc01 = wmma_bf16(a0.v, b1.v, acc01);
    acc10 = wmma_bf16(a1.v, b0.v, acc10);
    acc11 = wmma_bf16(a1.v, b1.v, acc11);
  }

  #pragma unroll
  for (int v = 0; v < 8; ++v) {
    const long mA = m0 + wm + v + 8 * hi;
    const long mB = mA + 16;
    C[mA * (long)N + n0 + wn + lm]      = acc00[v];
    C[mA * (long)N + n0 + wn + 16 + lm] = acc01[v];
    C[mB * (long)N + n0 + wn + lm]      = acc10[v];
    C[mB * (long)N + n0 + wn + 16 + lm] = acc11[v];
  }
}

// ---------------------------------------------------------------------------
// RMSNorm (row-wise), bf16 output
// ---------------------------------------------------------------------------
__global__ __launch_bounds__(256)
void rmsnorm_kernel(const float* __restrict__ src, int srcStride,
                    const float* __restrict__ w, __bf16* __restrict__ dst,
                    int dstStride, int n) {
  const int row = blockIdx.x;
  const float* x = src + (long)row * srcStride;
  float ss = 0.f;
  for (int i = threadIdx.x; i < n; i += blockDim.x) { float v = x[i]; ss += v * v; }
  __shared__ float red[8];
  for (int o = 16; o; o >>= 1) ss += __shfl_xor(ss, o, 32);
  if ((threadIdx.x & 31) == 0) red[threadIdx.x >> 5] = ss;
  __syncthreads();
  if (threadIdx.x < 8) {
    float t = red[threadIdx.x];
    for (int o = 4; o; o >>= 1) t += __shfl_xor(t, o, 8);
    if (threadIdx.x == 0) red[0] = t;
  }
  __syncthreads();
  const float inv = rsqrtf(red[0] / (float)n + EPS);
  for (int i = threadIdx.x; i < n; i += blockDim.x)
    dst[(long)row * dstStride + i] = to_bf16(x[i] * inv * w[i]);
}

// ---------------------------------------------------------------------------
// RoPE + assemble: qf [t][h][192] bf16, kf [t][h][192] bf16,
// vt [b][h][dv][s] bf16 (V transposed for the P*V B-fragment layout).
// ---------------------------------------------------------------------------
__global__ __launch_bounds__(256)
void assemble_kernel(const int* __restrict__ positions,
                     const float* __restrict__ qraw,   // [T][H*192]
                     const float* __restrict__ kvraw,  // [T][H*256]
                     const float* __restrict__ kvns,   // [T][576]
                     __bf16* __restrict__ qf, __bf16* __restrict__ kf,
                     __bf16* __restrict__ vt) {
  const int t = blockIdx.x;
  const int b = t / S_, sidx = t % S_;
  const float pos = (float)positions[t];

  for (int i = threadIdx.x; i < H_ * DN; i += blockDim.x) {
    const int h = i >> 7, d = i & 127;
    qf[(long)t * H_ * QKD + h * QKD + d] =
        to_bf16(qraw[(long)t * (H_ * QKD) + h * QKD + d]);
    kf[(long)t * H_ * QKD + h * QKD + d] =
        to_bf16(kvraw[(long)t * (H_ * (DN + DV)) + h * (DN + DV) + d]);
    vt[((long)(b * H_ + h) * DV + d) * S_ + sidx] =
        to_bf16(kvraw[(long)t * (H_ * (DN + DV)) + h * (DN + DV) + DN + d]);
  }
  for (int i = threadIdx.x; i < H_ * 32; i += blockDim.x) {
    const int h = i >> 5, fi = i & 31;
    const float freq = __powf(10000.0f, -(float)fi / 32.0f);
    float s, c; __sincosf(pos * freq, &s, &c);
    const float* qp = qraw + (long)t * (H_ * QKD) + h * QKD + DN;
    const float x1 = qp[2 * fi], x2 = qp[2 * fi + 1];
    __bf16* qd = qf + (long)t * H_ * QKD + h * QKD + DN;
    qd[2 * fi]     = to_bf16(x1 * c - x2 * s);
    qd[2 * fi + 1] = to_bf16(x2 * c + x1 * s);
  }
  for (int i = threadIdx.x; i < 32; i += blockDim.x) {
    const float freq = __powf(10000.0f, -(float)i / 32.0f);
    float s, c; __sincosf(pos * freq, &s, &c);
    const float x1 = kvns[(long)t * (KVR + DR) + KVR + 2 * i];
    const float x2 = kvns[(long)t * (KVR + DR) + KVR + 2 * i + 1];
    const float o1 = x1 * c - x2 * s, o2 = x2 * c + x1 * s;
    for (int h = 0; h < H_; ++h) {
      __bf16* kd = kf + (long)t * H_ * QKD + h * QKD + DN;
      kd[2 * i]     = to_bf16(o1);
      kd[2 * i + 1] = to_bf16(o2);
    }
  }
}

// ---------------------------------------------------------------------------
// Causal flash attention: block = (b,h, 128 q rows), 8 waves x 16 rows.
// K/V chunks (32 keys) double-buffered in LDS, shared by all waves.
// grid = (S/128, H, B), block = 256.
// ---------------------------------------------------------------------------
__global__ __launch_bounds__(256)
void attn_kernel(const __bf16* __restrict__ qf, const __bf16* __restrict__ kf,
                 const __bf16* __restrict__ vt, __bf16* __restrict__ attn) {
  __shared__ __attribute__((aligned(16))) __bf16 lK[2][32 * 200];   // [key][feat]
  __shared__ __attribute__((aligned(16))) __bf16 lV[2][128 * LDT];  // [dv][key]
  __shared__ __attribute__((aligned(16))) __bf16 lp[8][16 * LDT];

  const int b = blockIdx.z, h = blockIdx.y;
  const int wave = threadIdx.x >> 5, lane = threadIdx.x & 31;
  const int lm = lane & 15, hi = lane >> 4;
  const int qBlock = blockIdx.x * 128;
  const int qBase  = qBlock + wave * 16;

  const __bf16* Q    = qf + ((long)(b * S_ + qBase) * H_ + h) * QKD;
  const __bf16* Krow = kf + ((long)(b * S_) * H_ + h) * QKD;
  const __bf16* V    = vt + (long)(b * H_ + h) * DV * S_;

  // Q fragments: 6 chunks of K=32 over QKD=192, kept in registers
  frag qfr[6];
  #pragma unroll
  for (int c = 0; c < 6; ++c) {
    const __bf16* qrow = Q + (long)lm * (H_ * QKD) + c * 32;
    qfr[c].h2[0] = *(const v8bf*)(qrow + hi * 8);
    qfr[c].h2[1] = *(const v8bf*)(qrow + 16 + hi * 8);
  }

  auto stageKV = [&](int buf, int kk) {
    #pragma unroll
    for (int i = 0; i < 3; ++i) {
      const int idx = threadIdx.x + i * 256;
      const int r = idx / 24, c8 = (idx % 24) * 8;
      lds_cp16(&lK[buf][r * 200 + c8], Krow + (long)(kk + r) * (H_ * QKD) + c8);
    }
    #pragma unroll
    for (int i = 0; i < 2; ++i) {
      const int idx = threadIdx.x + i * 256;
      const int dv = idx >> 2, c8 = (idx & 3) * 8;
      lds_cp16(&lV[buf][dv * LDT + c8], V + (long)dv * S_ + kk + c8);
    }
  };

  float M_[8], L_[8];
  v8f O[8];
  #pragma unroll
  for (int v = 0; v < 8; ++v) { M_[v] = -3.0e38f; L_[v] = 0.f; O[v] = (v8f){}; }

  const int nChunks = (qBlock + 128) / 32;
  stageKV(0, 0);

  for (int ci = 0; ci < nChunks; ++ci) {
    const int kk = ci * 32;
    lds_cp_wait();
    __syncthreads();                 // chunk ci landed; chunk ci-1 reads retired
    if (ci + 1 < nChunks) stageKV((ci + 1) & 1, kk + 32);

    if (kk < qBase + 16) {   // wave-uniform causal skip (no barriers inside)
      const __bf16* bK = &lK[ci & 1][0];
      const __bf16* bV = &lV[ci & 1][0];
      // --- scores: two 16x16 key tiles ---
      v8f s0 = {}, s1 = {};
      #pragma unroll
      for (int c = 0; c < 6; ++c) {
        frag b0, b1;
        b0.h2[0] = *(const v8bf*)&bK[lm * 200 + c * 32 + hi * 8];
        b0.h2[1] = *(const v8bf*)&bK[lm * 200 + c * 32 + 16 + hi * 8];
        b1.h2[0] = *(const v8bf*)&bK[(16 + lm) * 200 + c * 32 + hi * 8];
        b1.h2[1] = *(const v8bf*)&bK[(16 + lm) * 200 + c * 32 + 16 + hi * 8];
        s0 = wmma_bf16(qfr[c].v, b0.v, s0);
        s1 = wmma_bf16(qfr[c].v, b1.v, s1);
      }
      // --- mask + online softmax (row = v + 8*hi, score col = lane lm) ---
      #pragma unroll
      for (int v = 0; v < 8; ++v) {
        const int qIdx = qBase + v + 8 * hi;
        float a0 = (kk + lm      <= qIdx) ? s0[v] * SCALE_ : -3.0e38f;
        float a1 = (kk + 16 + lm <= qIdx) ? s1[v] * SCALE_ : -3.0e38f;
        float mx = fmaxf(a0, a1);
        for (int o = 1; o < 16; o <<= 1) mx = fmaxf(mx, __shfl_xor(mx, o, 32));
        const float nm = fmaxf(M_[v], mx);
        const float al = __expf(M_[v] - nm);
        const float p0 = __expf(a0 - nm);
        const float p1 = __expf(a1 - nm);
        float rs = p0 + p1;
        for (int o = 1; o < 16; o <<= 1) rs += __shfl_xor(rs, o, 32);
        L_[v] = L_[v] * al + rs;
        M_[v] = nm;
        #pragma unroll
        for (int j = 0; j < 8; ++j) O[j][v] *= al;
        lp[wave][(v + 8 * hi) * LDT + lm]      = to_bf16(p0);
        lp[wave][(v + 8 * hi) * LDT + 16 + lm] = to_bf16(p1);
      }
      // --- reload P as a 16x32 A-fragment ---
      frag pf;
      pf.h2[0] = *(const v8bf*)&lp[wave][lm * LDT + hi * 8];
      pf.h2[1] = *(const v8bf*)&lp[wave][lm * LDT + 16 + hi * 8];
      // --- O += P * V over 8 dv column tiles ---
      #pragma unroll
      for (int j = 0; j < 8; ++j) {
        frag vfr;
        vfr.h2[0] = *(const v8bf*)&bV[(j * 16 + lm) * LDT + hi * 8];
        vfr.h2[1] = *(const v8bf*)&bV[(j * 16 + lm) * LDT + 16 + hi * 8];
        O[j] = wmma_bf16(pf.v, vfr.v, O[j]);
      }
    }
  }

  #pragma unroll
  for (int v = 0; v < 8; ++v) {
    const float inv = 1.0f / L_[v];
    const long t = (long)b * S_ + qBase + v + 8 * hi;
    #pragma unroll
    for (int j = 0; j < 8; ++j)
      attn[t * (H_ * DV) + h * DV + j * 16 + lm] = to_bf16(O[j][v] * inv);
  }
}

// ---------------------------------------------------------------------------
// Host-side orchestration
// ---------------------------------------------------------------------------
extern "C" void kernel_launch(void* const* d_in, const int* in_sizes, int n_in,
                              void* d_out, int out_size, void* d_ws, size_t ws_size,
                              hipStream_t stream) {
  const int*   positions = (const int*)  d_in[0];
  const float* hidden    = (const float*)d_in[1];
  const float* Wqa       = (const float*)d_in[2];
  const float* q_ln      = (const float*)d_in[3];
  const float* Wqb       = (const float*)d_in[4];
  const float* Wkva      = (const float*)d_in[5];
  const float* kv_ln     = (const float*)d_in[6];
  const float* Wkvb      = (const float*)d_in[7];
  const float* Wo        = (const float*)d_in[8];
  float* out = (float*)d_out;

  char* p = (char*)d_ws;
  auto alloc = [&](size_t bytes) -> void* {
    void* r = (void*)p;
    p += (bytes + 255) & ~(size_t)255;
    return r;
  };

  __bf16* hbf   = (__bf16*)alloc((size_t)T_ * HID * 2);
  __bf16* WqaT  = (__bf16*)alloc((size_t)HID * QLR * 2);               // [QLR][HID]
  __bf16* WqbT  = (__bf16*)alloc((size_t)QLR * (H_ * QKD) * 2);        // [H*QKD][QLR]
  __bf16* WkvaT = (__bf16*)alloc((size_t)HID * (KVR + DR) * 2);        // [KVR+DR][HID]
  __bf16* WkvbT = (__bf16*)alloc((size_t)KVR * (H_ * (DN + DV)) * 2);  // [H*256][KVR]
  __bf16* WoT   = (__bf16*)alloc((size_t)(H_ * DV) * HID * 2);         // [HID][H*DV]
  float*  ckq   = (float*) alloc((size_t)T_ * QLR * 4);
  __bf16* qc    = (__bf16*)alloc((size_t)T_ * QLR * 2);
  float*  qraw  = (float*) alloc((size_t)T_ * (H_ * QKD) * 4);
  float*  kvns  = (float*) alloc((size_t)T_ * (KVR + DR) * 4);
  __bf16* kvc   = (__bf16*)alloc((size_t)T_ * KVR * 2);
  float*  kvraw = (float*) alloc((size_t)T_ * (H_ * (DN + DV)) * 4);
  __bf16* qfb   = (__bf16*)alloc((size_t)T_ * H_ * QKD * 2);
  __bf16* kfb   = (__bf16*)alloc((size_t)T_ * H_ * QKD * 2);
  __bf16* vtb   = (__bf16*)alloc((size_t)B_ * H_ * DV * S_ * 2);
  __bf16* attnb = (__bf16*)alloc((size_t)T_ * (H_ * DV) * 2);

  // activations: straight convert; weights: convert + transpose to [N][K]
  cvt_bf16_kernel<<<2048, 256, 0, stream>>>(hidden, hbf, (long)T_ * HID);
  cvtT_bf16_kernel<<<dim3(QLR / 32, HID / 32), 256, 0, stream>>>(Wqa,  WqaT,  HID, QLR);
  cvtT_bf16_kernel<<<dim3((H_ * QKD) / 32, QLR / 32), 256, 0, stream>>>(Wqb, WqbT, QLR, H_ * QKD);
  cvtT_bf16_kernel<<<dim3((KVR + DR) / 32, HID / 32), 256, 0, stream>>>(Wkva, WkvaT, HID, KVR + DR);
  cvtT_bf16_kernel<<<dim3((H_ * (DN + DV)) / 32, KVR / 32), 256, 0, stream>>>(Wkvb, WkvbT, KVR, H_ * (DN + DV));
  cvtT_bf16_kernel<<<dim3(HID / 32, (H_ * DV) / 32), 256, 0, stream>>>(Wo, WoT, H_ * DV, HID);

  gemm_bf16_kernel<<<dim3(QLR / 64, T_ / 128), 256, 0, stream>>>(hbf, WqaT, ckq, T_, QLR, HID);
  gemm_bf16_kernel<<<dim3((KVR + DR) / 64, T_ / 128), 256, 0, stream>>>(hbf, WkvaT, kvns, T_, KVR + DR, HID);

  rmsnorm_kernel<<<T_, 256, 0, stream>>>(ckq,  QLR,      q_ln,  qc,  QLR, QLR);
  rmsnorm_kernel<<<T_, 256, 0, stream>>>(kvns, KVR + DR, kv_ln, kvc, KVR, KVR);

  gemm_bf16_kernel<<<dim3((H_ * QKD) / 64, T_ / 128), 256, 0, stream>>>(qc, WqbT, qraw, T_, H_ * QKD, QLR);
  gemm_bf16_kernel<<<dim3((H_ * (DN + DV)) / 64, T_ / 128), 256, 0, stream>>>(kvc, WkvbT, kvraw, T_, H_ * (DN + DV), KVR);

  assemble_kernel<<<T_, 256, 0, stream>>>(positions, qraw, kvraw, kvns, qfb, kfb, vtb);

  attn_kernel<<<dim3(S_ / 128, H_, B_), 256, 0, stream>>>(qfb, kfb, vtb, attnb);

  gemm_bf16_kernel<<<dim3(HID / 64, T_ / 128), 256, 0, stream>>>(attnb, WoT, out, T_, HID, H_ * DV);
}
